// ContrastiveLoss_6837587935562
// MI455X (gfx1250) — matrix operand order repaired and checked by reading, not compiled, not run on hardware
//
#include <hip/hip_runtime.h>

typedef __attribute__((ext_vector_type(16))) _Float16 v16h;
typedef __attribute__((ext_vector_type(8)))  _Float16 v8h;
typedef __attribute__((ext_vector_type(8)))  float    v8f;

constexpr int KCLS = 32;
constexpr int BS   = 64;
constexpr int F    = 128;
constexpr int N1   = KCLS * BS;   // 2048
constexpr int N    = 2 * N1;      // 4096
constexpr float TEMP = 0.01f;
constexpr float OW   = 0.5f;

constexpr int TILE = 16;
constexpr int NT   = N / TILE;            // 256 tiles per dimension
constexpr int WPB  = 8;                   // waves per block
constexpr int BLOCKS = NT / WPB;          // 32 blocks

// workspace layout (bytes)
constexpr size_t WS_FEATSH = 0;                              // N*F halves
constexpr size_t WS_LABELS = (size_t)N * F * 2;              // N ints
constexpr size_t WS_PART   = WS_LABELS + (size_t)N * 4;      // NT*4 floats

// ---------------------------------------------------------------------------
// Prep: fp32 -> f16 conversion of concat(feats1, feats2); label computation.
// ---------------------------------------------------------------------------
__global__ void ContrastivePrep(const float* __restrict__ f1,
                                const float* __restrict__ f2,
                                const int*   __restrict__ overlap,
                                _Float16*    __restrict__ featsH,
                                int*         __restrict__ labels) {
  int idx = blockIdx.x * blockDim.x + threadIdx.x;
  if (idx < N * F) {
    int row = idx / F;
    float v = (row < N1) ? f1[idx] : f2[idx - N1 * F];
    featsH[idx] = (_Float16)v;
  }
  if (idx < N) {
    int lab;
    if (idx < N1) {
      lab = idx / BS;
    } else {
      int c = (idx - N1) / BS;
      if (overlap[c]) {
        lab = c;
      } else {
        int excl = 0;
        for (int t = 0; t < c; ++t) excl += (overlap[t] ? 0 : 1);
        lab = KCLS + excl;
      }
    }
    labels[idx] = lab;
  }
}

// ---------------------------------------------------------------------------
// Main: fused WMMA GEMM + row reductions. One wave per 16-row tile.
// Pass 1: neg_sum, max_neg(logit), weighted positive logit sum.
// Pass 2: count positives whose logit beats max_neg.
// Epilogues are fully branchless (cndmask selects) so EXEC never changes and
// the VALU stream can co-issue with the XDL WMMA pipe.
// ---------------------------------------------------------------------------
__global__ __launch_bounds__(256) void ContrastiveMain(
    const _Float16* __restrict__ featsH,
    const int*      __restrict__ labels,
    float*          __restrict__ partials) {
  const int lane   = threadIdx.x & 31;
  const int wave   = threadIdx.x >> 5;
  const int tileI  = blockIdx.x * WPB + wave;
  const int half16 = lane >> 4;      // which 16-lane half
  const int nsub   = lane & 15;

  // A fragments: row M = nsub of tile I; 4 chunks of K=32.
  // 16-bit A 16x32 layout: halves[0..15] = K in [16*half16, +16) (contiguous, matches B).
  const _Float16* arow = featsH + (size_t)(tileI * TILE + nsub) * F;
  v16h a[4];
#pragma unroll
  for (int kk = 0; kk < 4; ++kk) {
    const v8h* p = (const v8h*)(arow + kk * 32 + half16 * 16);
    union { v16h v; v8h h[2]; } u;
    u.h[0] = p[0];   // K base
    u.h[1] = p[1];   // K base + 8
    a[kk] = u.v;
  }

  int li[8];
#pragma unroll
  for (int v = 0; v < 8; ++v) li[v] = labels[tileI * TILE + v + 8 * half16];
  const bool is2I = (tileI * TILE) >= N1;

  float negsum[8], maxneg[8], wsum[8], swl[8], poscnt[8];
#pragma unroll
  for (int v = 0; v < 8; ++v) {
    negsum[v] = 0.0f; maxneg[v] = -1e30f;
    wsum[v] = 0.0f; swl[v] = 0.0f; poscnt[v] = 0.0f;
  }

  // ---------------- Pass 1 ----------------
  for (int tj = 0; tj < NT; ++tj) {
    const _Float16* brow = featsH + (size_t)(tj * TILE + nsub) * F;
    if (tj + 1 < NT) __builtin_prefetch((const void*)(brow + TILE * F), 0, 3);
    v8f c = {};
#pragma unroll
    for (int kk = 0; kk < 4; ++kk) {
      // 16-bit B 32x16 layout: contiguous 16 halves of column's source row.
      v16h b = *(const v16h*)(brow + kk * 32 + half16 * 16);
      c = __builtin_amdgcn_wmma_f32_16x16x32_f16(false, a[kk], false, b,
                                                 (short)0, c, false, false);
    }
    const int  j    = tj * TILE + nsub;     // column index (fixed per lane)
    const int  lj   = labels[j];
    const bool is2J = (tj * TILE) >= N1;
    const float w   = (is2I != is2J) ? OW : 1.0f;
#pragma unroll
    for (int v = 0; v < 8; ++v) {
      const int   i     = tileI * TILE + v + 8 * half16;
      const float logit = c[v] * TEMP;
      const float ex    = __expf(logit);          // computed unconditionally
      const bool  neg   = (li[v] != lj);
      const bool  pos   = (!neg) & (i != j);
      negsum[v] += neg ? ex : 0.0f;
      maxneg[v]  = fmaxf(maxneg[v], neg ? logit : -1e30f);
      const float wp = pos ? w : 0.0f;
      wsum[v]   += wp;
      swl[v]     = fmaf(wp, logit, swl[v]);
      poscnt[v] += pos ? 1.0f : 0.0f;
    }
  }

  // Reduce across the 16 lanes that share each row (xor butterfly, half-wave).
#pragma unroll
  for (int v = 0; v < 8; ++v) {
#pragma unroll
    for (int m = 1; m <= 8; m <<= 1) {
      negsum[v] += __shfl_xor(negsum[v], m, 32);
      maxneg[v]  = fmaxf(maxneg[v], __shfl_xor(maxneg[v], m, 32));
      wsum[v]   += __shfl_xor(wsum[v], m, 32);
      swl[v]    += __shfl_xor(swl[v], m, 32);
      poscnt[v] += __shfl_xor(poscnt[v], m, 32);
    }
  }

  // ---------------- Pass 2: accuracy counting (branchless) ----------------
  float correct = 0.0f;
  for (int tj = 0; tj < NT; ++tj) {
    const _Float16* brow = featsH + (size_t)(tj * TILE + nsub) * F;
    if (tj + 1 < NT) __builtin_prefetch((const void*)(brow + TILE * F), 0, 3);
    v8f c = {};
#pragma unroll
    for (int kk = 0; kk < 4; ++kk) {
      v16h b = *(const v16h*)(brow + kk * 32 + half16 * 16);
      c = __builtin_amdgcn_wmma_f32_16x16x32_f16(false, a[kk], false, b,
                                                 (short)0, c, false, false);
    }
    const int j  = tj * TILE + nsub;
    const int lj = labels[j];
#pragma unroll
    for (int v = 0; v < 8; ++v) {
      const int   i     = tileI * TILE + v + 8 * half16;
      const float logit = c[v] * TEMP;
      const bool  hit   = (li[v] == lj) & (i != j) & (logit > maxneg[v]);
      correct += hit ? 1.0f : 0.0f;
    }
  }
#pragma unroll
  for (int m = 1; m <= 16; m <<= 1) correct += __shfl_xor(correct, m, 32);

  // Per-tile partials: loss = sum_rows [log(neg_sum)*W - S], pos count.
  float lp = 0.0f, pc = 0.0f;
#pragma unroll
  for (int v = 0; v < 8; ++v) {
    lp += __logf(negsum[v]) * wsum[v] - swl[v];
    pc += poscnt[v];
  }
  lp += __shfl_xor(lp, 16, 32);   // combine the two half-wave row groups
  pc += __shfl_xor(pc, 16, 32);

  if (lane == 0) {
    partials[tileI * 4 + 0] = lp;
    partials[tileI * 4 + 1] = correct;
    partials[tileI * 4 + 2] = pc;
  }
}

// ---------------------------------------------------------------------------
// Finish: reduce 256 tile partials -> (acc, loss).
// ---------------------------------------------------------------------------
__global__ void ContrastiveFinish(const float* __restrict__ partials,
                                  float* __restrict__ out) {
  __shared__ float s0[NT], s1[NT], s2[NT];
  const int t = threadIdx.x;
  s0[t] = partials[t * 4 + 0];
  s1[t] = partials[t * 4 + 1];
  s2[t] = partials[t * 4 + 2];
  __syncthreads();
  for (int step = NT / 2; step > 0; step >>= 1) {
    if (t < step) {
      s0[t] += s0[t + step];
      s1[t] += s1[t + step];
      s2[t] += s2[t + step];
    }
    __syncthreads();
  }
  if (t == 0) {
    out[0] = s1[0] / s2[0];   // acc
    out[1] = s0[0] / s2[0];   // loss
  }
}

extern "C" void kernel_launch(void* const* d_in, const int* in_sizes, int n_in,
                              void* d_out, int out_size, void* d_ws, size_t ws_size,
                              hipStream_t stream) {
  const float* f1      = (const float*)d_in[0];
  const float* f2      = (const float*)d_in[1];
  const int*   overlap = (const int*)d_in[2];
  (void)in_sizes; (void)n_in; (void)out_size; (void)ws_size;

  char* ws = (char*)d_ws;
  _Float16* featsH   = (_Float16*)(ws + WS_FEATSH);
  int*      labels   = (int*)(ws + WS_LABELS);
  float*    partials = (float*)(ws + WS_PART);
  float*    out      = (float*)d_out;

  ContrastivePrep<<<(N * F + 255) / 256, 256, 0, stream>>>(f1, f2, overlap,
                                                           featsH, labels);
  ContrastiveMain<<<BLOCKS, 256, 0, stream>>>(featsH, labels, partials);
  ContrastiveFinish<<<1, NT, 0, stream>>>(partials, out);
}
